// ST_GCN_83442624626950
// MI455X (gfx1250) — matrix operand order repaired and checked by reading, not compiled, not run on hardware
//
#include <hip/hip_runtime.h>
#include <hip/hip_bf16.h>
#include <stdint.h>

// ---------------------------------------------------------------------------
// ST-GCN inference for MI455X (gfx1250): all matmuls via v_wmma_f32_16x16x32_f16
// Activations stored channels-last f16:  X[n][v][t][c]   (n = 64 sequences)
// Tiles staged global->LDS with the CDNA5 async path (ASYNCcnt) + double buffer.
// ---------------------------------------------------------------------------

typedef __attribute__((ext_vector_type(16))) _Float16 v16h;
typedef __attribute__((ext_vector_type(8)))  float    v8f;

union FragA { v16h v; uint32_t u[8]; };
union FragC { v8f  v; float    f[8]; };

#define GBM 128
#define GBN 64
#define GBK 32

#define NSEQ 64
#define NV   21

#if defined(__HIP_DEVICE_COMPILE__) && \
    __has_builtin(__builtin_amdgcn_global_load_async_to_lds_b128) && \
    __has_builtin(__builtin_amdgcn_s_wait_asynccnt)
#define USE_ASYNC_LDS 1
#else
#define USE_ASYNC_LDS 0
#endif

#if USE_ASYNC_LDS
typedef int v4i_gcc __attribute__((vector_size(16)));
typedef __attribute__((address_space(1))) v4i_gcc* async_gptr_t;
typedef __attribute__((address_space(3))) v4i_gcc* async_lptr_t;
#endif

// 16-byte global -> LDS copy (async on gfx1250 toolchains that expose it)
__device__ __forceinline__ void cp16(const void* g, void* l) {
#if USE_ASYNC_LDS
  void* gg = const_cast<void*>(g);
  __builtin_amdgcn_global_load_async_to_lds_b128(
      (async_gptr_t)gg, (async_lptr_t)l, 0, 0);
#else
  *reinterpret_cast<uint4*>(l) = *reinterpret_cast<const uint4*>(g);
#endif
}

__device__ __forceinline__ void wait_async_lds() {
#if USE_ASYNC_LDS
  __builtin_amdgcn_s_wait_asynccnt(0);
#endif
}

// ---------------------------------------------------------------------------
// Small helper kernels
// ---------------------------------------------------------------------------

__global__ void adj_kernel(float* __restrict__ Anorm) {
  const int ea[19] = {5,17,18,19,5,14,15,16,5,11,12,13,5,8,9,10,5,6,7};
  const int eb[19] = {17,18,19,4,14,15,16,3,11,12,13,2,8,9,10,1,6,7,0};
  __shared__ float A[441];
  __shared__ float dinv[21];
  int t = threadIdx.x;
  for (int i = t; i < 441; i += blockDim.x) A[i] = 0.f;
  __syncthreads();
  if (t == 0) {
    for (int e = 0; e < 19; e++) {
      A[ea[e]*21 + eb[e]] = 1.f;
      A[eb[e]*21 + ea[e]] = 1.f;
    }
  }
  __syncthreads();
  if (t < 21) {
    float s = 0.f;
    for (int j = 0; j < 21; j++) s += A[t*21 + j];
    dinv[t] = rsqrtf(s + 0.001f);
  }
  __syncthreads();
  for (int i = t; i < 441; i += blockDim.x) {
    int r = i / 21, c = i % 21;
    Anorm[i] = dinv[r] * A[i] * dinv[c];
  }
}

// x: (32,3,300,21,2) f32 -> X0[n=b*2+m][v][t][c] f16 with BN0
__global__ void bn0_kernel(const float* __restrict__ x, _Float16* __restrict__ X0,
                           const float* __restrict__ g, const float* __restrict__ be,
                           const float* __restrict__ mn, const float* __restrict__ vr) {
  int idx = blockIdx.x * 256 + threadIdx.x;
  const int total = NSEQ * NV * 300 * 3;
  if (idx >= total) return;
  int c = idx % 3;
  int t = (idx / 3) % 300;
  int v = (idx / 900) % 21;
  int n = idx / (900 * 21);
  int b = n >> 1, m = n & 1;
  float xv = x[((((long)b*3 + c)*300 + t)*21 + v)*2 + m];
  float s = g[c] * rsqrtf(vr[c] + 1e-5f);
  X0[idx] = (_Float16)((xv - mn[c]) * s + be[c]);
}

// Fold biases/BN into per-channel epilogue constants.
// epi layout: [0]=bias_sp, [256]=s1, [512]=sh1, [768]=s2, [1024]=sh2
__global__ void prep_epi_kernel(const float* __restrict__ bnb, const float* __restrict__ bself,
                                const float* __restrict__ g1, const float* __restrict__ be1,
                                const float* __restrict__ m1, const float* __restrict__ v1,
                                const float* __restrict__ g2, const float* __restrict__ be2,
                                const float* __restrict__ m2, const float* __restrict__ v2,
                                float* __restrict__ epi, int Co) {
  int o = threadIdx.x;
  if (o >= Co) return;
  epi[o] = bnb[o] + bself[o];
  float s1 = g1[o] * rsqrtf(v1[o] + 1e-5f);
  epi[256 + o]  = s1;
  epi[512 + o]  = be1[o] - m1[o] * s1;
  float s2 = g2[o] * rsqrtf(v2[o] + 1e-5f);
  epi[768 + o]  = s2;
  epi[1024 + o] = be2[o] - m2[o] * s2;
}

// Weight repack into pair-interleaved f16:  dword(kk,o) = { B[2kk][o], B[2kk+1][o] }
// so GEMM B-tile staging is a straight 16B async copy.

// spatial: rows k<C -> w_nb[o][k], rows C..2C -> w_self[o][k-C]
__global__ void repack_sp_kernel(const float* __restrict__ wnb, const float* __restrict__ wself,
                                 _Float16* __restrict__ B, int C, int Co) {
  int idx = blockIdx.x * 256 + threadIdx.x;
  if (idx >= 2 * C * Co) return;
  int o = idx % Co, k = idx / Co;
  float v = (k < C) ? wnb[(long)o*C + k] : wself[(long)o*C + (k - C)];
  B[(((long)(k >> 1))*Co + o)*2 + (k & 1)] = (_Float16)v;
}

// temporal: row kg = tap*C + c  ->  w_temp[o][c][tap]
__global__ void repack_tm_kernel(const float* __restrict__ wt, _Float16* __restrict__ B, int C) {
  int idx = blockIdx.x * 256 + threadIdx.x;
  if (idx >= 9 * C * C) return;
  int o  = idx % C;
  int kg = idx / C;           // 0 .. 9C-1
  int c  = kg % C;
  int k  = kg / C;
  float v = wt[((long)o*C + c)*9 + k];
  B[(((long)(kg >> 1))*C + o)*2 + (kg & 1)] = (_Float16)v;
}

// residual: row k -> w_res[o][k]; rows k>=C zero-filled (odd-K padding)
__global__ void repack_res_kernel(const float* __restrict__ wr, _Float16* __restrict__ B,
                                  int C, int Co) {
  int Kpad = (C + 1) & ~1;
  int idx = blockIdx.x * 256 + threadIdx.x;
  if (idx >= Kpad * Co) return;
  int o = idx % Co, k = idx / Co;
  float v = (k < C) ? wr[(long)o*C + k] : 0.f;
  B[(((long)(k >> 1))*Co + o)*2 + (k & 1)] = (_Float16)v;
}

// Z[n][u][t][0:C]   = sum_v (Anorm*M)[v][u] * X[n][v][t][c]
// Z[n][u][t][C:2C]  = M[u][u] * X[n][u][t][c]
__global__ void aggregate_kernel(const _Float16* __restrict__ X, _Float16* __restrict__ Z,
                                 const float* __restrict__ Anorm, const float* __restrict__ Mm,
                                 int C, int T) {
  __shared__ float AM[441];
  __shared__ float MD[21];
  for (int i = threadIdx.x; i < 441; i += 256) AM[i] = Anorm[i] * Mm[i];
  if (threadIdx.x < 21) MD[threadIdx.x] = Mm[threadIdx.x * 22];
  __syncthreads();
  long idx = (long)blockIdx.x * 256 + threadIdx.x;
  long total = (long)NSEQ * NV * T * C;
  if (idx >= total) return;
  int c = (int)(idx % C);
  long q = idx / C;
  int t = (int)(q % T); q /= T;
  int u = (int)(q % NV);
  int n = (int)(q / NV);
  float acc = 0.f;
  #pragma unroll
  for (int v = 0; v < 21; v++)
    acc += AM[v*21 + u] * (float)X[(((long)n*NV + v)*T + t)*C + c];
  float sf = MD[u] * (float)X[(((long)n*NV + u)*T + t)*C + c];
  long zb = (((long)n*NV + u)*T + t) * (2L * C);
  Z[zb + c]     = (_Float16)acc;
  Z[zb + C + c] = (_Float16)sf;
}

// mean over (m,v,t) of X9[n][v][t=75][c=256]
__global__ void pool_kernel(const _Float16* __restrict__ X9, float* __restrict__ feat) {
  int idx = blockIdx.x * 256 + threadIdx.x;
  if (idx >= 32 * 256) return;
  int c = idx & 255, b = idx >> 8;
  float s = 0.f;
  for (int m = 0; m < 2; m++)
    for (int v = 0; v < 21; v++)
      for (int t = 0; t < 75; t++)
        s += (float)X9[((((long)(b*2 + m)*NV + v)*75) + t)*256 + c];
  feat[idx] = s * (1.f / (2.f * 21.f * 75.f));
}

__global__ void fc_kernel(const float* __restrict__ feat, const float* __restrict__ fcw,
                          const float* __restrict__ fcb, float* __restrict__ out) {
  int idx = blockIdx.x * 256 + threadIdx.x;
  if (idx >= 32 * 60) return;
  int j = idx % 60, b = idx / 60;
  float s = fcb[j];
  for (int c = 0; c < 256; c++) s += feat[b*256 + c] * fcw[j*256 + c];
  out[idx] = s;
}

// ---------------------------------------------------------------------------
// Generic WMMA GEMM:  D[r][o] = sum_k A[rowaddr(r)+k] * B[k][o]
// rowaddr(r) = (r / tBlk)*blkStride + (r % tBlk)*rowStride
// B is pair-interleaved: dword (kk, o) holds rows {2kk, 2kk+1} at column o.
// mode 0: f16 out:  y = relu(acc+bias)*scale + shift     (spatial conv + BN1)
// mode 1: f32 out:  y = acc + bias                        (residual projection)
// ---------------------------------------------------------------------------
__global__ __launch_bounds__(256) void gemm_f16_kernel(
    const _Float16* __restrict__ A, const uint32_t* __restrict__ Bp,
    void* __restrict__ D, int R, int N, int K,
    int tBlk, long blkStride, int rowStride,
    const float* __restrict__ bias, const float* __restrict__ scale,
    const float* __restrict__ shift, int mode) {
  __shared__ __align__(16) _Float16 lA[2][GBM][GBK];
  __shared__ __align__(16) uint32_t lB[2][GBK/2][GBN];

  const int tid  = threadIdx.x;
  const int lane = tid & 31;
  const int m16  = lane & 15;
  const int hif  = lane >> 4;
  const int wave = tid >> 5;
  const int wm   = (wave >> 1) * 32;   // 4 wave rows
  const int wn   = (wave & 1) * 32;    // 2 wave cols
  const int rowBase = blockIdx.x * GBM;
  const int colBase = blockIdx.y * GBN;
  const int Kpair = (K + 1) >> 1;

  FragC acc[2][2];
  #pragma unroll
  for (int a = 0; a < 2; a++)
    #pragma unroll
    for (int b = 0; b < 2; b++)
      #pragma unroll
      for (int r = 0; r < 8; r++) acc[a][b].f[r] = 0.f;

  // A staging geometry: row = tid>>1, two 8-elem (16B) chunks per thread
  const int ar  = tid >> 1;
  const int agr = rowBase + ar;
  const bool arvalid = agr < R;
  long arbase = 0;
  if (arvalid)
    arbase = (long)(agr / tBlk) * blkStride + (long)(agr % tBlk) * rowStride;
  // B staging geometry: pair-row = tid>>4, four dwords (16B) per thread
  const int bkk  = tid >> 4;           // 0..15
  const int bcol = (tid & 15) * 4;     // 0..60

  auto stage = [&](int buf, int kb) {
    #pragma unroll
    for (int q = 0; q < 2; q++) {
      int kc = (tid & 1) * 16 + q * 8;
      if (arvalid && (kb + kc + 8 <= K)) {
        cp16(A + arbase + kb + kc, &lA[buf][ar][kc]);
      } else {
        #pragma unroll
        for (int e = 0; e < 8; e++) {
          int k = kb + kc + e;
          bool ok = arvalid && (k < K);
          _Float16 v = A[ok ? (arbase + k) : 0];
          lA[buf][ar][kc + e] = ok ? v : (_Float16)0.f;
        }
      }
    }
    int prow = (kb >> 1) + bkk;
    if (prow < Kpair) {
      cp16(Bp + (size_t)prow * N + colBase + bcol, &lB[buf][bkk][bcol]);
    } else {
      *reinterpret_cast<uint4*>(&lB[buf][bkk][bcol]) = make_uint4(0u, 0u, 0u, 0u);
    }
  };

  int buf = 0;
  stage(0, 0);
  wait_async_lds();
  __syncthreads();

  for (int kb = 0; kb < K; kb += GBK) {
    int nkb = kb + GBK;
    if (nkb < K) stage(buf ^ 1, nkb);

    // ---- fragments (contiguous k = 16*hif + s, same in A and B) + 2x2 WMMA
    FragA af[2], bf[2];
    #pragma unroll
    for (int mi = 0; mi < 2; mi++) {
      const uint32_t* arow = reinterpret_cast<const uint32_t*>(&lA[buf][wm + mi*16 + m16][0]);
      #pragma unroll
      for (int j = 0; j < 8; j++)
        af[mi].u[j] = arow[(hif << 3) + j];
    }
    #pragma unroll
    for (int ni = 0; ni < 2; ni++) {
      int col = wn + ni*16 + m16;
      #pragma unroll
      for (int j = 0; j < 8; j++)
        bf[ni].u[j] = lB[buf][(hif << 3) + j][col];
    }
    #pragma unroll
    for (int mi = 0; mi < 2; mi++)
      #pragma unroll
      for (int ni = 0; ni < 2; ni++)
        acc[mi][ni].v = __builtin_amdgcn_wmma_f32_16x16x32_f16(
            false, af[mi].v, false, bf[ni].v, (short)0, acc[mi][ni].v, false, false);

    wait_async_lds();
    __syncthreads();
    buf ^= 1;
  }

  // ---- epilogue
  #pragma unroll
  for (int ni = 0; ni < 2; ni++) {
    int col = colBase + wn + ni*16 + m16;
    if (col >= N) continue;
    float bs = bias ? bias[col] : 0.f;
    float sc = (mode == 0) ? scale[col] : 1.f;
    float sh = (mode == 0) ? shift[col] : 0.f;
    #pragma unroll
    for (int mi = 0; mi < 2; mi++) {
      #pragma unroll
      for (int r = 0; r < 8; r++) {
        int row = rowBase + wm + mi*16 + hif*8 + r;
        if (row >= R) continue;
        float y = acc[mi][ni].f[r] + bs;
        if (mode == 0) {
          y = fmaxf(y, 0.f);
          y = y * sc + sh;
          ((_Float16*)D)[(long)row*N + col] = (_Float16)y;
        } else {
          ((float*)D)[(long)row*N + col] = y;
        }
      }
    }
  }
}

// ---------------------------------------------------------------------------
// Temporal conv (kernel 9, pad 4, stride s) as sliding-window GEMM + fused
// bias/ReLU/BN2/residual-add/ReLU.  One block per (t-tile, c-tile, n*v).
// H layout [nv][Tin][C];  Bp pair-interleaved [9C/2][C];  O layout [nv][Tout][C].
// ---------------------------------------------------------------------------
__global__ __launch_bounds__(256) void tconv_f16_kernel(
    const _Float16* __restrict__ H, const uint32_t* __restrict__ Bp,
    _Float16* __restrict__ O, const void* __restrict__ resPtr, int resF32,
    int Tin, int Tout, int C, int stride,
    const float* __restrict__ btemp, const float* __restrict__ s2,
    const float* __restrict__ sh2) {
  __shared__ __align__(16) _Float16 lA[2][GBM][GBK];
  __shared__ __align__(16) uint32_t lB[2][GBK/2][GBN];

  const int tid  = threadIdx.x;
  const int lane = tid & 31;
  const int m16  = lane & 15;
  const int hif  = lane >> 4;
  const int wave = tid >> 5;
  const int wm   = (wave >> 1) * 32;
  const int wn   = (wave & 1) * 32;
  const int rowBase = blockIdx.x * GBM;
  const int colBase = blockIdx.y * GBN;
  const int nv = blockIdx.z;
  const int K  = 9 * C;
  const int Kpair = K >> 1;
  const long hbase = (long)nv * Tin * C;

  FragC acc[2][2];
  #pragma unroll
  for (int a = 0; a < 2; a++)
    #pragma unroll
    for (int b = 0; b < 2; b++)
      #pragma unroll
      for (int r = 0; r < 8; r++) acc[a][b].f[r] = 0.f;

  const int ar = tid >> 1;
  const int t_out = rowBase + ar;
  const int bkk  = tid >> 4;
  const int bcol = (tid & 15) * 4;

  auto stage = [&](int buf, int kb) {
    int tap  = kb / C;
    int c0b  = kb - tap * C;           // K-steps stay inside one tap (C % 32 == 0)
    int t_in = t_out * stride + tap - 4;
    bool ok = (t_out < Tout) && (t_in >= 0) && (t_in < Tin);
    #pragma unroll
    for (int q = 0; q < 2; q++) {
      int kc = (tid & 1) * 16 + q * 8;
      if (ok) {
        cp16(H + hbase + (long)t_in*C + c0b + kc, &lA[buf][ar][kc]);
      } else {
        *reinterpret_cast<uint4*>(&lA[buf][ar][kc]) = make_uint4(0u, 0u, 0u, 0u);
      }
    }
    int prow = (kb >> 1) + bkk;
    if (prow < Kpair) {
      cp16(Bp + (size_t)prow * C + colBase + bcol, &lB[buf][bkk][bcol]);
    } else {
      *reinterpret_cast<uint4*>(&lB[buf][bkk][bcol]) = make_uint4(0u, 0u, 0u, 0u);
    }
  };

  int buf = 0;
  stage(0, 0);
  wait_async_lds();
  __syncthreads();

  for (int kb = 0; kb < K; kb += GBK) {
    int nkb = kb + GBK;
    if (nkb < K) stage(buf ^ 1, nkb);

    FragA af[2], bf[2];
    #pragma unroll
    for (int mi = 0; mi < 2; mi++) {
      const uint32_t* arow = reinterpret_cast<const uint32_t*>(&lA[buf][wm + mi*16 + m16][0]);
      #pragma unroll
      for (int j = 0; j < 8; j++)
        af[mi].u[j] = arow[(hif << 3) + j];
    }
    #pragma unroll
    for (int ni = 0; ni < 2; ni++) {
      int col = wn + ni*16 + m16;
      #pragma unroll
      for (int j = 0; j < 8; j++)
        bf[ni].u[j] = lB[buf][(hif << 3) + j][col];
    }
    #pragma unroll
    for (int mi = 0; mi < 2; mi++)
      #pragma unroll
      for (int ni = 0; ni < 2; ni++)
        acc[mi][ni].v = __builtin_amdgcn_wmma_f32_16x16x32_f16(
            false, af[mi].v, false, bf[ni].v, (short)0, acc[mi][ni].v, false, false);

    wait_async_lds();
    __syncthreads();
    buf ^= 1;
  }

  const long obase = (long)nv * Tout * C;
  #pragma unroll
  for (int ni = 0; ni < 2; ni++) {
    int col = colBase + wn + ni*16 + m16;
    if (col >= C) continue;
    float bt = btemp[col], sc = s2[col], sh = sh2[col];
    #pragma unroll
    for (int mi = 0; mi < 2; mi++) {
      #pragma unroll
      for (int r = 0; r < 8; r++) {
        int to = rowBase + wm + mi*16 + hif*8 + r;
        if (to >= Tout) continue;
        long oi = obase + (long)to*C + col;
        float y = fmaxf(acc[mi][ni].f[r] + bt, 0.f);
        y = y * sc + sh;
        float rv = resF32 ? ((const float*)resPtr)[oi]
                          : (float)((const _Float16*)resPtr)[oi];
        y = fmaxf(y + rv, 0.f);
        O[oi] = (_Float16)y;
      }
    }
  }
}

// ---------------------------------------------------------------------------
// Host launcher
// ---------------------------------------------------------------------------
extern "C" void kernel_launch(void* const* d_in, const int* in_sizes, int n_in,
                              void* d_out, int out_size, void* d_ws, size_t ws_size,
                              hipStream_t stream) {
  (void)in_sizes; (void)n_in; (void)out_size;

  static const int IN_CH[9]  = {3, 64, 64, 64, 128, 128, 128, 256, 256};
  static const int OUT_CH[9] = {64, 64, 64, 128, 128, 128, 256, 256, 256};
  static const int STR[9]    = {1, 1, 1, 2, 1, 1, 2, 1, 1};
  static const int T_IN[9]   = {300, 300, 300, 300, 150, 150, 150, 75, 75};

  int p = 0;
  const float* x    = (const float*)d_in[p++];
  const float* bn0g = (const float*)d_in[p++];
  const float* bn0b = (const float*)d_in[p++];
  const float* bn0m = (const float*)d_in[p++];
  const float* bn0v = (const float*)d_in[p++];

  struct UnitP {
    const float *M, *wnb, *bnb, *wself, *bself, *wtemp, *btemp;
    const float *g1, *be1, *m1, *v1, *g2, *be2, *m2, *v2;
    const float *wres, *bres;
  } up[9];
  for (int i = 0; i < 9; i++) {
    up[i].M     = (const float*)d_in[p++];
    up[i].wnb   = (const float*)d_in[p++];
    up[i].bnb   = (const float*)d_in[p++];
    up[i].wself = (const float*)d_in[p++];
    up[i].bself = (const float*)d_in[p++];
    up[i].wtemp = (const float*)d_in[p++];
    up[i].btemp = (const float*)d_in[p++];
    up[i].g1 = (const float*)d_in[p++];
    up[i].be1 = (const float*)d_in[p++];
    up[i].m1 = (const float*)d_in[p++];
    up[i].v1 = (const float*)d_in[p++];
    up[i].g2 = (const float*)d_in[p++];
    up[i].be2 = (const float*)d_in[p++];
    up[i].m2 = (const float*)d_in[p++];
    up[i].v2 = (const float*)d_in[p++];
    if (IN_CH[i] != OUT_CH[i]) {
      up[i].wres = (const float*)d_in[p++];
      up[i].bres = (const float*)d_in[p++];
    } else {
      up[i].wres = nullptr;
      up[i].bres = nullptr;
    }
  }
  const float* fcw = (const float*)d_in[p++];
  const float* fcb = (const float*)d_in[p++];

  // ---- workspace carve (aligned 256B)
  uintptr_t w0 = (uintptr_t)d_ws;
  uintptr_t w  = w0;
  auto take = [&](size_t n) -> void* {
    void* r = (void*)w;
    w += (n + 255) & ~(size_t)255;
    return r;
  };
  float*    Anorm = (float*)take(441 * sizeof(float));
  float*    epi   = (float*)take(5 * 256 * sizeof(float));
  _Float16* Bsp   = (_Float16*)take((size_t)512 * 256 * 2);
  _Float16* Btm   = (_Float16*)take((size_t)2304 * 256 * 2);
  _Float16* Brs   = (_Float16*)take((size_t)256 * 256 * 2);
  float*    feat  = (float*)take((size_t)32 * 256 * sizeof(float));
  const size_t XBYTES = (size_t)NSEQ * NV * 300 * 64 * 2;   // 51.6 MB
  const size_t ZBYTES = (size_t)NSEQ * NV * 300 * 128 * 2;  // 103.2 MB
  _Float16* Xa = (_Float16*)take(XBYTES);
  _Float16* Xb = (_Float16*)take(XBYTES);
  void*     ZR = take(ZBYTES);   // Z (f16) and residual (f32) alias, sequential use
  _Float16* H1 = (_Float16*)take(ZBYTES);
  if (w - w0 > ws_size) return;  // insufficient scratch: do nothing (deterministic)

  _Float16* Xbuf[2] = {Xa, Xb};

  adj_kernel<<<1, 256, 0, stream>>>(Anorm);
  {
    int total = NSEQ * NV * 300 * 3;
    bn0_kernel<<<(total + 255) / 256, 256, 0, stream>>>(x, Xa, bn0g, bn0b, bn0m, bn0v);
  }

  for (int i = 0; i < 9; i++) {
    const int C  = IN_CH[i];
    const int Co = OUT_CH[i];
    const int Ti = T_IN[i];
    const int s  = STR[i];
    const int To = Ti / s;
    _Float16* Xin  = Xbuf[i & 1];
    _Float16* Xout = Xbuf[(i + 1) & 1];

    prep_epi_kernel<<<1, 256, 0, stream>>>(up[i].bnb, up[i].bself,
        up[i].g1, up[i].be1, up[i].m1, up[i].v1,
        up[i].g2, up[i].be2, up[i].m2, up[i].v2, epi, Co);

    {
      int tot = 2 * C * Co;
      repack_sp_kernel<<<(tot + 255) / 256, 256, 0, stream>>>(up[i].wnb, up[i].wself, Bsp, C, Co);
    }
    {
      int tot = 9 * Co * Co;
      repack_tm_kernel<<<(tot + 255) / 256, 256, 0, stream>>>(up[i].wtemp, Btm, Co);
    }

    {
      long tot = (long)NSEQ * NV * Ti * C;
      aggregate_kernel<<<(unsigned)((tot + 255) / 256), 256, 0, stream>>>(
          Xin, (_Float16*)ZR, Anorm, up[i].M, C, Ti);
    }

    // spatial GEMM:  Z [R x 2C]  x  Bsp [2C x Co]  ->  H1  (relu + BN1, f16)
    {
      int R = NSEQ * NV * Ti;
      dim3 grid((R + GBM - 1) / GBM, Co / GBN);
      gemm_f16_kernel<<<grid, 256, 0, stream>>>(
          (const _Float16*)ZR, (const uint32_t*)Bsp, H1, R, Co, 2 * C,
          R, 0L, 2 * C, epi, epi + 256, epi + 512, 0);
    }

    const void* resp;
    int rf32;
    if (up[i].wres) {
      int Kpad = (C + 1) & ~1;
      int tot = Kpad * Co;
      repack_res_kernel<<<(tot + 255) / 256, 256, 0, stream>>>(up[i].wres, Brs, C, Co);
      int Rr = NSEQ * NV * To;
      dim3 grid((Rr + GBM - 1) / GBM, Co / GBN);
      gemm_f16_kernel<<<grid, 256, 0, stream>>>(
          Xin, (const uint32_t*)Brs, ZR, Rr, Co, C,
          To, (long)Ti * C, s * C, up[i].bres, nullptr, nullptr, 1);
      resp = ZR; rf32 = 1;
    } else {
      resp = Xin; rf32 = 0;
    }

    // temporal conv GEMM + fused epilogue -> Xout
    {
      dim3 grid((To + GBM - 1) / GBM, Co / GBN, NSEQ * NV);
      tconv_f16_kernel<<<grid, 256, 0, stream>>>(
          H1, (const uint32_t*)Btm, Xout, resp, rf32, Ti, To, Co, s,
          up[i].btemp, epi + 768, epi + 1024);
    }
  }

  pool_kernel<<<(32 * 256 + 255) / 256, 256, 0, stream>>>(Xbuf[1], feat);
  fc_kernel<<<(32 * 60 + 255) / 256, 256, 0, stream>>>(feat, fcw, fcb, (float*)d_out);
}
